// AttentionDecoderLayer_6554120093817
// MI455X (gfx1250) — compile-verified, matrix operand order checked by reading, not run on hardware
//
#include <hip/hip_runtime.h>
#include <hip/hip_bf16.h>

// ---------------------------------------------------------------------------
// Transformer decoder layer on gfx1250 (MI455X), bf16 WMMA + flash attention.
// B=2, S=2048, d=1024, H=16, hd=64, ff=4096.  ~137 GFLOP, ~150 MB HBM traffic.
// GEMMs: 128x128 block tile, 64x64 per wave, LDS double-buffered staging via
// GLOBAL_LOAD_ASYNC_TO_LDS (ASYNCcnt) when available.
// ---------------------------------------------------------------------------

#define BATCH 2
#define SEQ   2048
#define DMODEL 1024
#define NH    16
#define HD    64
#define FF    4096
#define MROWS (BATCH * SEQ) // 4096

typedef __attribute__((ext_vector_type(16))) __bf16 v16bf;
typedef __attribute__((ext_vector_type(8)))  float  v8f;
typedef __attribute__((ext_vector_type(4)))  int    v4i;

union Frag { unsigned u[8]; v16bf b; };

#if defined(__has_builtin)
#if __has_builtin(__builtin_amdgcn_global_load_async_to_lds_b128)
#define HAVE_ASYNC_LDS 1
#endif
#endif
#ifndef HAVE_ASYNC_LDS
#define HAVE_ASYNC_LDS 0
#endif

__device__ __forceinline__ unsigned short f2bf(float f) {
  unsigned u = __float_as_uint(f);
  return (unsigned short)((u + 0x7FFFu + ((u >> 16) & 1u)) >> 16); // RNE
}

// 16B global -> LDS copy: async (ASYNCcnt) when toolchain has the builtin.
// Builtin signature (from clang diagnostic): (int4 AS1*, int4 AS3*, Ii, Ii).
__device__ __forceinline__ void cp16_g2l(const unsigned short* g, unsigned short* l) {
#if HAVE_ASYNC_LDS
  typedef __attribute__((address_space(1))) v4i gv4i;
  typedef __attribute__((address_space(3))) v4i lv4i;
  __builtin_amdgcn_global_load_async_to_lds_b128((gv4i*)g, (lv4i*)l, 0, 0);
#else
  struct alignas(16) U4 { unsigned a, b, c, d; };
  *(U4*)l = *(const U4*)g;
#endif
}

__device__ __forceinline__ void async_wait0() {
#if HAVE_ASYNC_LDS
#if __has_builtin(__builtin_amdgcn_s_wait_asynccnt)
  __builtin_amdgcn_s_wait_asynccnt(0);
#else
  asm volatile("s_wait_asynccnt 0x0" ::: "memory");
#endif
#endif
}

// 16-bit A-matrix 16x32 fragment (ISA 7.12.2): lane holds row M=lane&15,
// half=lane>>4; dword map u[v] = p32[h*4 + (v&3) + (v>=4 ? 8 : 0)].
// B fragments use the same loader on pre-transposed weights (column of B =
// contiguous row of B^T along K).
__device__ __forceinline__ v16bf load_frag_k32(const unsigned short* p, int hf) {
  const unsigned* p32 = (const unsigned*)p;
  Frag f;
  int b0 = hf * 4;
#pragma unroll
  for (int v = 0; v < 4; ++v) f.u[v] = p32[b0 + v];
#pragma unroll
  for (int v = 0; v < 4; ++v) f.u[4 + v] = p32[b0 + 8 + v];
  return f.b;
}

__device__ __forceinline__ v8f wmma_bf16(v16bf a, v16bf b, v8f c) {
  return __builtin_amdgcn_wmma_f32_16x16x32_bf16(false, a, false, b, (short)0, c,
                                                 false, false);
}

// ---------------------------------------------------------------------------
// Weight cast + transpose: W[K][N] fp32 -> WT[N][K] bf16
// ---------------------------------------------------------------------------
__global__ __launch_bounds__(256) void cast_transpose_k(
    const float* __restrict__ W, unsigned short* __restrict__ WT, int K, int N) {
  size_t idx = (size_t)blockIdx.x * 256 + threadIdx.x;
  if (idx >= (size_t)K * N) return;
  int k = (int)(idx / N), n = (int)(idx % N);
  WT[(size_t)n * K + k] = f2bf(W[idx]);
}

// ---------------------------------------------------------------------------
// LayerNorm row kernel: x fp32 [rows][D] -> bf16 out
// ---------------------------------------------------------------------------
__global__ __launch_bounds__(256) void layernorm_k(
    const float* __restrict__ x, const float* __restrict__ g,
    const float* __restrict__ be, unsigned short* __restrict__ out, int D) {
  __shared__ float s1[8], s2[8];
  int row = blockIdx.x;
  const float* xr = x + (size_t)row * D;
  float sum = 0.f, sq = 0.f;
  for (int i = threadIdx.x; i < D; i += 256) { float v = xr[i]; sum += v; sq += v * v; }
#pragma unroll
  for (int m = 16; m; m >>= 1) { sum += __shfl_xor(sum, m, 32); sq += __shfl_xor(sq, m, 32); }
  int w = threadIdx.x >> 5;
  if ((threadIdx.x & 31) == 0) { s1[w] = sum; s2[w] = sq; }
  __syncthreads();
  float ts = 0.f, tq = 0.f;
#pragma unroll
  for (int i = 0; i < 8; ++i) { ts += s1[i]; tq += s2[i]; }
  float mu = ts / D;
  float var = tq / D - mu * mu;
  float rstd = rsqrtf(var + 1e-5f);
  for (int i = threadIdx.x; i < D; i += 256)
    out[(size_t)row * D + i] = f2bf((xr[i] - mu) * rstd * g[i] + be[i]);
}

// ---------------------------------------------------------------------------
// GEMM: C[M][N] = A[M][K](bf16) @ B (via BT[N][K] bf16) + bias (+relu)(+resid)
// 128 threads = 4 waves (2x2); block tile 128x128; wave tile 64x64 = 16 WMMA
// accumulators.  K staged in 32-deep double-buffered LDS tiles (async copies).
// Arithmetic intensity: 64 flop per L2 byte, 32 flop per LDS byte read.
// ---------------------------------------------------------------------------
template <bool RELU, bool RESID, bool OUTBF>
__global__ __launch_bounds__(128) void gemm_k(
    const unsigned short* __restrict__ A, const unsigned short* __restrict__ BT,
    const float* __restrict__ bias, const float* __restrict__ resid,
    float* __restrict__ Cf, unsigned short* __restrict__ Cb, int M, int N, int K) {
  __shared__ __align__(16) unsigned short Abuf[2][128 * 32];
  __shared__ __align__(16) unsigned short Bbuf[2][128 * 32];

  int t = threadIdx.x;
  int lane = t & 31;
  int wave = t >> 5;
  int r = lane & 15, hf = lane >> 4;
  int wm = (wave >> 1) * 64, wn = (wave & 1) * 64;

  const unsigned short* gA = A + (size_t)blockIdx.y * 128 * K;
  const unsigned short* gB = BT + (size_t)blockIdx.x * 128 * K;

  // coalesced staging map: 512 16B chunks, 4 per thread; groups of 4 lanes
  // fetch consecutive 16B of one 64B row.
  int srow = t >> 2, sch = (t & 3) * 8; // +32 rows per i

  v8f acc[4][4] = {};

  auto stage = [&](int buf, int k0) {
#pragma unroll
    for (int i = 0; i < 4; ++i) {
      int row = srow + 32 * i;
      cp16_g2l(gA + (size_t)row * K + k0 + sch, &Abuf[buf][row * 32 + sch]);
      cp16_g2l(gB + (size_t)row * K + k0 + sch, &Bbuf[buf][row * 32 + sch]);
    }
  };

  stage(0, 0);
  async_wait0();
  __syncthreads();

  int nIter = K / 32;
  for (int it = 0; it < nIter; ++it) {
    if (it + 1 < nIter) stage((it + 1) & 1, (it + 1) * 32);

    const unsigned short* Ab = &Abuf[it & 1][0];
    const unsigned short* Bb = &Bbuf[it & 1][0];
    v16bf af[4];
#pragma unroll
    for (int i = 0; i < 4; ++i)
      af[i] = load_frag_k32(Ab + (wm + i * 16 + r) * 32, hf);
#pragma unroll
    for (int j = 0; j < 4; ++j) {
      v16bf bfr = load_frag_k32(Bb + (wn + j * 16 + r) * 32, hf);
#pragma unroll
      for (int i = 0; i < 4; ++i)
        acc[i][j] = wmma_bf16(af[i], bfr, acc[i][j]);
    }

    if (it + 1 < nIter) {
      async_wait0();
      __syncthreads();
    }
  }

#pragma unroll
  for (int i = 0; i < 4; ++i)
#pragma unroll
    for (int j = 0; j < 4; ++j) {
      int col = blockIdx.x * 128 + wn + j * 16 + r;
      float bv = bias[col];
#pragma unroll
      for (int e = 0; e < 8; ++e) {
        int row = blockIdx.y * 128 + wm + i * 16 + hf * 8 + e;
        float v = acc[i][j][e] + bv;
        if (RELU) v = fmaxf(v, 0.f);
        if (RESID) v += resid[(size_t)row * N + col];
        if (OUTBF) Cb[(size_t)row * N + col] = f2bf(v);
        else       Cf[(size_t)row * N + col] = v;
      }
    }
}

// ---------------------------------------------------------------------------
// RoPE (interleaved pairs) fp32 -> bf16, optional 1/sqrt(hd) scale.
// ---------------------------------------------------------------------------
__global__ __launch_bounds__(256) void rope_k(
    const float* __restrict__ src, unsigned short* __restrict__ dst, float scale) {
  int idx = blockIdx.x * 256 + threadIdx.x;
  if (idx >= BATCH * SEQ * NH * (HD / 2)) return;
  int p = idx & 31;
  int s = (idx >> 9) & (SEQ - 1);
  float freq = __powf(10000.0f, -(float)p / 32.0f);
  float sn, cs;
  __sincosf((float)s * freq, &sn, &cs);
  size_t base = (size_t)idx * 2;
  float t0 = src[base], t1 = src[base + 1];
  dst[base]     = f2bf((t0 * cs - t1 * sn) * scale);
  dst[base + 1] = f2bf((t0 * sn + t1 * cs) * scale);
}

// ---------------------------------------------------------------------------
// V transpose per head: v fp32 [B,S,H,hd] -> vt bf16 [B,H,hd,S]
// ---------------------------------------------------------------------------
__global__ __launch_bounds__(256) void vtrans_k(
    const float* __restrict__ v, unsigned short* __restrict__ vt) {
  int idx = blockIdx.x * 256 + threadIdx.x;
  if (idx >= BATCH * NH * HD * SEQ) return;
  int s = idx & (SEQ - 1);
  int d = (idx >> 11) & (HD - 1);
  int h = (idx >> 17) & (NH - 1);
  int b = idx >> 21;
  vt[idx] = f2bf(v[((size_t)(b * SEQ + s) * NH + h) * HD + d]);
}

// ---------------------------------------------------------------------------
// Flash attention: 4 waves/block, each wave owns a 16-row Q tile for one
// (b,h); 32-key blocks, online softmax, never materializes S x S scores.
// ---------------------------------------------------------------------------
__global__ __launch_bounds__(128) void flash_attn_k(
    const unsigned short* __restrict__ qB, const unsigned short* __restrict__ kB,
    const unsigned short* __restrict__ vT, unsigned short* __restrict__ ctx) {
  __shared__ unsigned short Pl[4][16 * 32];
  int lane = threadIdx.x & 31;
  int wave = threadIdx.x >> 5;
  int r = lane & 15, hf = lane >> 4;
  int bh = blockIdx.y;
  int b = bh >> 4, h = bh & 15;
  int qt = blockIdx.x * 4 + wave;

  const unsigned short* qrow = qB + ((size_t)(b * SEQ + qt * 16 + r) * DMODEL + h * HD);
  v16bf qf0 = load_frag_k32(qrow, hf);
  v16bf qf1 = load_frag_k32(qrow + 32, hf);

  v8f o[4] = {};
  float m_run[8], l_run[8];
#pragma unroll
  for (int j = 0; j < 8; ++j) { m_run[j] = -1e30f; l_run[j] = 0.f; }

  unsigned short* P = Pl[wave];
  const unsigned short* vbase = vT + (size_t)bh * HD * SEQ;

  for (int kb0 = 0; kb0 < SEQ; kb0 += 32) {
    const unsigned short* kr0 = kB + ((size_t)(b * SEQ + kb0 + r) * DMODEL + h * HD);
    const unsigned short* kr1 = kB + ((size_t)(b * SEQ + kb0 + 16 + r) * DMODEL + h * HD);
    if (kb0 + 32 < SEQ) { // hint next K tile toward the caches
      __builtin_prefetch(kr0 + 32 * DMODEL, 0, 1);
      __builtin_prefetch(kr1 + 32 * DMODEL, 0, 1);
    }
    v16bf k00 = load_frag_k32(kr0, hf);
    v16bf k01 = load_frag_k32(kr0 + 32, hf);
    v16bf k10 = load_frag_k32(kr1, hf);
    v16bf k11 = load_frag_k32(kr1 + 32, hf);
    v8f s0 = {}, s1 = {};
    s0 = wmma_bf16(qf0, k00, s0);
    s0 = wmma_bf16(qf1, k01, s0);
    s1 = wmma_bf16(qf0, k10, s1);
    s1 = wmma_bf16(qf1, k11, s1);

    float scale[8];
#pragma unroll
    for (int j = 0; j < 8; ++j) {
      float mx = fmaxf(s0[j], s1[j]);
#pragma unroll
      for (int d = 1; d < 16; d <<= 1) mx = fmaxf(mx, __shfl_xor(mx, d, 32));
      float mn = fmaxf(m_run[j], mx);
      float p0 = __expf(s0[j] - mn);
      float p1 = __expf(s1[j] - mn);
      float ps = p0 + p1;
#pragma unroll
      for (int d = 1; d < 16; d <<= 1) ps += __shfl_xor(ps, d, 32);
      float corr = __expf(m_run[j] - mn);
      l_run[j] = l_run[j] * corr + ps;
      m_run[j] = mn;
      scale[j] = corr;
      P[(hf * 8 + j) * 32 + r]      = f2bf(p0);
      P[(hf * 8 + j) * 32 + 16 + r] = f2bf(p1);
    }
#pragma unroll
    for (int t = 0; t < 4; ++t)
#pragma unroll
      for (int j = 0; j < 8; ++j) o[t][j] *= scale[j];

    v16bf pf;
    {
      const unsigned* l32 = (const unsigned*)(P + r * 32);
      Frag f;
      int b0 = hf * 4;
#pragma unroll
      for (int v = 0; v < 4; ++v) f.u[v] = l32[b0 + v];
#pragma unroll
      for (int v = 0; v < 4; ++v) f.u[4 + v] = l32[b0 + 8 + v];
      pf = f.b;
    }

#pragma unroll
    for (int t = 0; t < 4; ++t) {
      const unsigned short* vrow = vbase + (size_t)(t * 16 + r) * SEQ + kb0;
      v16bf vf = load_frag_k32(vrow, hf);
      o[t] = wmma_bf16(pf, vf, o[t]);
    }
  }

#pragma unroll
  for (int j = 0; j < 8; ++j) {
    float inv = 1.0f / l_run[j];
    int row = qt * 16 + hf * 8 + j;
#pragma unroll
    for (int t = 0; t < 4; ++t)
      ctx[(size_t)(b * SEQ + row) * DMODEL + h * HD + t * 16 + r] = f2bf(o[t][j] * inv);
  }
}

// ---------------------------------------------------------------------------
// Launch: requires ~136 MB workspace.
// ---------------------------------------------------------------------------
extern "C" void kernel_launch(void* const* d_in, const int* in_sizes, int n_in,
                              void* d_out, int out_size, void* d_ws, size_t ws_size,
                              hipStream_t stream) {
  (void)in_sizes; (void)n_in; (void)out_size; (void)ws_size;
  const float* x   = (const float*)d_in[0];
  const float* Wq  = (const float*)d_in[1];
  const float* bq  = (const float*)d_in[2];
  const float* Wk  = (const float*)d_in[3];
  const float* bk  = (const float*)d_in[4];
  const float* Wv  = (const float*)d_in[5];
  const float* bv  = (const float*)d_in[6];
  const float* Wo  = (const float*)d_in[7];
  const float* bo  = (const float*)d_in[8];
  const float* W1  = (const float*)d_in[9];
  const float* b1  = (const float*)d_in[10];
  const float* W2  = (const float*)d_in[11];
  const float* b2  = (const float*)d_in[12];
  const float* g1  = (const float*)d_in[13];
  const float* be1 = (const float*)d_in[14];
  const float* g2  = (const float*)d_in[15];
  const float* be2 = (const float*)d_in[16];

  char* ws = (char*)d_ws;
  size_t off = 0;
  auto alloc = [&](size_t bytes) {
    char* p = ws + off;
    off += (bytes + 255) & ~(size_t)255;
    return p;
  };
  const size_t DD = (size_t)DMODEL * DMODEL;
  unsigned short* wqT = (unsigned short*)alloc(DD * 2);
  unsigned short* wkT = (unsigned short*)alloc(DD * 2);
  unsigned short* wvT = (unsigned short*)alloc(DD * 2);
  unsigned short* woT = (unsigned short*)alloc(DD * 2);
  unsigned short* w1T = (unsigned short*)alloc((size_t)DMODEL * FF * 2);
  unsigned short* w2T = (unsigned short*)alloc((size_t)FF * DMODEL * 2);
  unsigned short* xnb = (unsigned short*)alloc((size_t)MROWS * DMODEL * 2);
  float* qf = (float*)alloc((size_t)MROWS * DMODEL * 4);
  float* kf = (float*)alloc((size_t)MROWS * DMODEL * 4); // qf+kf reused as h1 later
  float* vf = (float*)alloc((size_t)MROWS * DMODEL * 4);
  unsigned short* qB  = (unsigned short*)alloc((size_t)MROWS * DMODEL * 2);
  unsigned short* kB  = (unsigned short*)alloc((size_t)MROWS * DMODEL * 2);
  unsigned short* vT  = (unsigned short*)alloc((size_t)MROWS * DMODEL * 2);
  unsigned short* ctx = (unsigned short*)alloc((size_t)MROWS * DMODEL * 2);
  float* x1 = (float*)alloc((size_t)MROWS * DMODEL * 4);
  unsigned short* xn2 = (unsigned short*)alloc((size_t)MROWS * DMODEL * 2);
  unsigned short* h1 = (unsigned short*)qf; // 32 MB alias over dead qf+kf

  auto nb = [](size_t n) { return (unsigned)((n + 255) / 256); };

  cast_transpose_k<<<nb(DD), 256, 0, stream>>>(Wq, wqT, DMODEL, DMODEL);
  cast_transpose_k<<<nb(DD), 256, 0, stream>>>(Wk, wkT, DMODEL, DMODEL);
  cast_transpose_k<<<nb(DD), 256, 0, stream>>>(Wv, wvT, DMODEL, DMODEL);
  cast_transpose_k<<<nb(DD), 256, 0, stream>>>(Wo, woT, DMODEL, DMODEL);
  cast_transpose_k<<<nb((size_t)DMODEL * FF), 256, 0, stream>>>(W1, w1T, DMODEL, FF);
  cast_transpose_k<<<nb((size_t)FF * DMODEL), 256, 0, stream>>>(W2, w2T, FF, DMODEL);

  layernorm_k<<<MROWS, 256, 0, stream>>>(x, g1, be1, xnb, DMODEL);

  dim3 gsq(DMODEL / 128, MROWS / 128); // (8, 32)
  gemm_k<false, false, false><<<gsq, 128, 0, stream>>>(xnb, wqT, bq, nullptr, qf, nullptr, MROWS, DMODEL, DMODEL);
  gemm_k<false, false, false><<<gsq, 128, 0, stream>>>(xnb, wkT, bk, nullptr, kf, nullptr, MROWS, DMODEL, DMODEL);
  gemm_k<false, false, false><<<gsq, 128, 0, stream>>>(xnb, wvT, bv, nullptr, vf, nullptr, MROWS, DMODEL, DMODEL);

  unsigned npair = BATCH * SEQ * NH * (HD / 2);
  rope_k<<<nb(npair), 256, 0, stream>>>(qf, qB, 0.125f); // 1/sqrt(64)
  rope_k<<<nb(npair), 256, 0, stream>>>(kf, kB, 1.0f);
  vtrans_k<<<nb((size_t)BATCH * NH * HD * SEQ), 256, 0, stream>>>(vf, vT);

  dim3 gfa(SEQ / 64, BATCH * NH);
  flash_attn_k<<<gfa, 128, 0, stream>>>(qB, kB, vT, ctx);

  gemm_k<false, true, false><<<gsq, 128, 0, stream>>>(ctx, woT, bo, x, x1, nullptr, MROWS, DMODEL, DMODEL);

  layernorm_k<<<MROWS, 256, 0, stream>>>(x1, g2, be2, xn2, DMODEL);

  dim3 gff1(FF / 128, MROWS / 128); // (32, 32)
  gemm_k<true, false, true><<<gff1, 128, 0, stream>>>(xn2, w1T, b1, nullptr, nullptr, h1, MROWS, FF, DMODEL);

  gemm_k<false, true, false><<<gsq, 128, 0, stream>>>(h1, w2T, b2, x1, (float*)d_out, nullptr, MROWS, DMODEL, FF);
}
